// GRU_Layer_55808805044507
// MI455X (gfx1250) — matrix operand order repaired and checked by reading, not compiled, not run on hardware
//
#include <hip/hip_runtime.h>
#include <hip/hip_bf16.h>

// ---------------------------------------------------------------------------
// GRU layer for MI455X (gfx1250): B=256, D=128, T=2048
// ---------------------------------------------------------------------------

#define D_DIM   128
#define T_SEQ   2048
#define B_BATCH 256
#define N3      384          // 3*D
#define NTILES  24           // N3/16
#define KBLK    4            // D/32

typedef __attribute__((ext_vector_type(16))) __bf16 v16bf;
typedef __attribute__((ext_vector_type(8)))  float  v8f;

#if defined(__has_builtin)
#if __has_builtin(__builtin_amdgcn_global_load_async_to_lds_b128) && \
    __has_builtin(__builtin_amdgcn_s_wait_asynccnt)
#define GRU_ASYNC 1
#endif
#endif
#ifndef GRU_ASYNC
#define GRU_ASYNC 0
#endif

#if GRU_ASYNC
typedef int v4i_ __attribute__((vector_size(16)));
typedef __attribute__((address_space(1))) v4i_ gv4i;   // global
typedef __attribute__((address_space(3))) v4i_ lv4i;   // LDS
#endif

// float -> bf16 (round-to-nearest-even)
__device__ __forceinline__ unsigned short f2bf(float f) {
    union { float f; unsigned u; } v; v.f = f;
    unsigned r = v.u + 0x7FFFu + ((v.u >> 16) & 1u);
    return (unsigned short)(r >> 16);
}

// 16-bit A-matrix 16x32 element mapping (ISA 7.12.2)
__device__ __forceinline__ int a_k_of(int lane, int i) {
    return (lane < 16) ? ((i < 8) ? i : i + 8)
                       : ((i < 8) ? i + 8 : i + 16);
}
// 16-bit B-matrix 32x16: same K order as A (lanes 0-15 K=0..7,16..23; lanes 16-31 K=8..15,24..31).
__device__ __forceinline__ int b_k_of(int lane, int i) {
    return (lane < 16) ? ((i < 8) ? i : i + 8)
                       : ((i < 8) ? i + 8 : i + 16);
}

__device__ __forceinline__ float sigmoidf_(float x) {
    return 1.0f / (1.0f + __expf(-x));
}

// ---------------------------------------------------------------------------
// k0: swizzle W,U (fp32 [128,384]) into bf16 WMMA B-fragment order:
//     frag[j][kb][lane][i],  j=N-tile, kb=K-block(32), 16 bf16 per lane.
// ---------------------------------------------------------------------------
__global__ void k0_swizzle_weights(const float* __restrict__ W,
                                   const float* __restrict__ U,
                                   unsigned short* __restrict__ Wf,
                                   unsigned short* __restrict__ Uf) {
    int idx = blockIdx.x * 256 + threadIdx.x;        // 0 .. 49151
    int i    = idx & 15;
    int lane = (idx >> 4) & 31;
    int kb   = (idx >> 9) & 3;
    int j    = idx >> 11;
    int K = kb * 32 + b_k_of(lane, i);
    int N = j * 16 + (lane & 15);
    Wf[idx] = f2bf(W[K * N3 + N]);
    Uf[idx] = f2bf(U[K * N3 + N]);
}

// ---------------------------------------------------------------------------
// k1: GX[t,b,n] = x[b,:,t] @ W + bias   (stored [T, B, 384] fp32)
//     grid = (T/16, B), 128 threads (4 waves), M-tile = 16 timesteps.
// ---------------------------------------------------------------------------
__global__ void k1_gx_gemm(const float* __restrict__ x,
                           const unsigned short* __restrict__ Wf,
                           const float* __restrict__ bias,
                           float* __restrict__ GX) {
    __shared__ __align__(32) float          sX[D_DIM * 16];     // [K=d][M=t]
    __shared__ __align__(32) unsigned short sA[KBLK * 32 * 16]; // A-frags

    const int tid  = threadIdx.x;
    const int lane = tid & 31;
    const int wave = tid >> 5;
    const int t0 = blockIdx.x * 16;
    const int b  = blockIdx.y;

    // load x[b, d, t0:t0+16] -> sX[d][m]  (16 contiguous floats per d)
    {
        const int d = tid;  // 128 threads
        const float4* src = (const float4*)(x + (size_t)b * D_DIM * T_SEQ
                                              + (size_t)d * T_SEQ + t0);
        float4 r0 = src[0], r1 = src[1], r2 = src[2], r3 = src[3];
        float* row = &sX[d * 16];
        *(float4*)(row + 0)  = r0; *(float4*)(row + 4)  = r1;
        *(float4*)(row + 8)  = r2; *(float4*)(row + 12) = r3;
    }
    __syncthreads();

    // pack bf16 A-fragments: 128 threads -> (kb, lane), 16 elems each
    {
        int pkb = tid >> 5;
        int pln = tid & 31;
        int M = pln & 15;
        unsigned short* dst = &sA[(pkb * 32 + pln) * 16];
        #pragma unroll
        for (int i = 0; i < 16; ++i) {
            int K = pkb * 32 + a_k_of(pln, i);
            dst[i] = f2bf(sX[K * 16 + M]);
        }
    }
    __syncthreads();

    v16bf a[KBLK];
    #pragma unroll
    for (int kb = 0; kb < KBLK; ++kb)
        a[kb] = *reinterpret_cast<const v16bf*>(&sA[(kb * 32 + lane) * 16]);

    #pragma unroll
    for (int jj = 0; jj < 6; ++jj) {
        const int j = wave * 6 + jj;
        v8f acc = {};
        #pragma unroll
        for (int kb = 0; kb < KBLK; ++kb) {
            v16bf bfr = *reinterpret_cast<const v16bf*>(
                &Wf[((j * KBLK + kb) * 32 + lane) * 16]);
            acc = __builtin_amdgcn_wmma_f32_16x16x32_bf16(
                false, a[kb], false, bfr, (short)0, acc, false, false);
        }
        const int n  = j * 16 + (lane & 15);
        const float bv = bias[n];
        #pragma unroll
        for (int r = 0; r < 8; ++r) {
            int m = (lane < 16) ? r : r + 8;
            GX[((size_t)(t0 + m) * B_BATCH + b) * N3 + n] = acc[r] + bv;
        }
    }
}

// ---------------------------------------------------------------------------
// k2: serial recurrence. 16 WGs (one per 16-row batch tile), 256 thr (8 waves)
//     U resident in LDS (96KB bf16 frags). h lives in registers; the gate
//     phase writes bf16(h_new) straight into the A-fragment LDS slots.
//     GX tiles double-buffered in LDS via async global->LDS loads (ASYNCcnt).
// ---------------------------------------------------------------------------
__global__ void k2_recurrence(const float* __restrict__ GX,
                              const unsigned short* __restrict__ Uf,
                              float* __restrict__ H) {
    __shared__ __align__(32) unsigned short sU[NTILES * KBLK * 32 * 16]; // 96KB
    __shared__ __align__(32) float          sGH[16 * N3];                // 24KB
    __shared__ __align__(32) unsigned short sA[KBLK * 32 * 16];          //  4KB
#if GRU_ASYNC
    __shared__ __align__(16) float          sGX[2 * 16 * N3];            // 48KB
#endif

    const int tid  = threadIdx.x;
    const int lane = tid & 31;
    const int wave = tid >> 5;
    const int b0 = blockIdx.x * 16;

    // stage U fragments into LDS (128-bit copies)
    {
        const uint4* src = (const uint4*)Uf;
        uint4* dst = (uint4*)sU;
        const int n16 = (NTILES * KBLK * 32 * 16) / 8;   // 6144 uint4
        for (int i = tid; i < n16; i += 256) dst[i] = src[i];
    }
    // h0 = 0 -> A fragments are all zero
    #pragma unroll
    for (int e = 0; e < 8; ++e) sA[tid + 256 * e] = 0;

    // ---- per-thread elementwise constants (invariant over t) ----
    int   slotE[8];     // A-fragment slot for this (br,d)
    int   gofE[8];      // br*384 + d  (index into GX row block and sGH)
    int   hofE[8];      // (b0+br)*128 + d (H store offset within one step)
    float hreg[8];
    #pragma unroll
    for (int e = 0; e < 8; ++e) {
        int flat = tid + 256 * e;          // 0..2047
        int d  = flat & (D_DIM - 1);
        int br = flat >> 7;
        int kb = d >> 5;
        int kl = d & 31;
        int g  = kl >> 3;                  // which 8-group of the 32-K block
        int ln = br + ((g & 1) << 4);
        int i  = (kl & 7) + ((g >= 2) ? 8 : 0);
        slotE[e] = (kb * 32 + ln) * 16 + i;
        gofE[e]  = br * N3 + d;
        hofE[e]  = (b0 + br) * D_DIM + d;
        hreg[e]  = 0.0f;
    }

    // per-wave WMMA store constants
    const int mbase = (lane < 16) ? 0 : 8;
    const int ncol  = lane & 15;

#if GRU_ASYNC
    // preload GX tile for t=0 into buffer 0 (6 x b128 per thread)
    {
        const float* g0 = GX + (size_t)b0 * N3;
        #pragma unroll
        for (int q = 0; q < 6; ++q) {
            int c = (tid + 256 * q) * 4;   // float index, 16B chunks
            __builtin_amdgcn_global_load_async_to_lds_b128(
                (gv4i*)(g0 + c), (lv4i*)(&sGX[c]), 0, 0);
        }
    }
#endif
    __syncthreads();

    for (int t = 0; t < T_SEQ; ++t) {
#if GRU_ASYNC
        if (t + 1 < T_SEQ) {
            const float* gn = GX + ((size_t)(t + 1) * B_BATCH + b0) * N3;
            float* dbuf = &sGX[((t + 1) & 1) * 16 * N3];
            #pragma unroll
            for (int q = 0; q < 6; ++q) {
                int c = (tid + 256 * q) * 4;
                __builtin_amdgcn_global_load_async_to_lds_b128(
                    (gv4i*)(gn + c), (lv4i*)(&dbuf[c]), 0, 0);
            }
        }
#else
        if (t + 1 < T_SEQ) {
            const float* nx = GX + ((size_t)(t + 1) * B_BATCH + b0) * N3;
            __builtin_prefetch(nx + tid * 24, 0, 3);
        }
#endif

        // gh = h @ U : wave w owns N-tiles j = 3w .. 3w+2
        v16bf a[KBLK];
        #pragma unroll
        for (int kb = 0; kb < KBLK; ++kb)
            a[kb] = *reinterpret_cast<const v16bf*>(&sA[(kb * 32 + lane) * 16]);

        #pragma unroll
        for (int jj = 0; jj < 3; ++jj) {
            const int j = wave * 3 + jj;
            v8f acc = {};
            #pragma unroll
            for (int kb = 0; kb < KBLK; ++kb) {
                v16bf bfr = *reinterpret_cast<const v16bf*>(
                    &sU[((j * KBLK + kb) * 32 + lane) * 16]);
                acc = __builtin_amdgcn_wmma_f32_16x16x32_bf16(
                    false, a[kb], false, bfr, (short)0, acc, false, false);
            }
            const int n = j * 16 + ncol;
            #pragma unroll
            for (int r = 0; r < 8; ++r)
                sGH[(mbase + r) * N3 + n] = acc[r];
        }

#if GRU_ASYNC
        // retire previous buffer's async loads (6 newer ones may remain)
        if (t + 1 < T_SEQ) __builtin_amdgcn_s_wait_asynccnt(6);
        else               __builtin_amdgcn_s_wait_asynccnt(0);
#endif
        __syncthreads();

#if GRU_ASYNC
        const float* gsrc = &sGX[(t & 1) * 16 * N3];
#else
        const float* gsrc = GX + ((size_t)t * B_BATCH + b0) * N3;
#endif
        float* Ht = H + (size_t)t * B_BATCH * D_DIM;

        // gates + state update; write bf16(h_new) directly into A-frag slot
        #pragma unroll
        for (int e = 0; e < 8; ++e) {
            int go = gofE[e];
            float xz = gsrc[go];
            float xr = gsrc[go + D_DIM];
            float xn = gsrc[go + 2 * D_DIM];
            float hz = sGH[go];
            float hr = sGH[go + D_DIM];
            float hn = sGH[go + 2 * D_DIM];
            float z = sigmoidf_(xz + hz);
            float r = sigmoidf_(xr + hr);
            float y = xn + r * hn;
            float n = 2.0f * sigmoidf_(2.0f * y) - 1.0f;   // tanh(y)
            float hnew = z * hreg[e] + (1.0f - z) * n;
            hreg[e] = hnew;
            sA[slotE[e]] = f2bf(hnew);
            Ht[hofE[e]] = hnew;
        }
        __syncthreads();
    }
}

// ---------------------------------------------------------------------------
// k3: out[b,d,t] = H[t,b,d]   (LDS-tiled transpose, coalesced both sides)
// ---------------------------------------------------------------------------
__global__ void k3_transpose(const float* __restrict__ H,
                             float* __restrict__ out) {
    __shared__ float tile[32][33];
    const int tid = threadIdx.x;
    const int tx = tid & 31;
    const int ty = tid >> 5;          // 0..7
    const int t0 = blockIdx.x * 32;
    const int d0 = blockIdx.y * 32;
    const int b  = blockIdx.z;

    #pragma unroll
    for (int ii = 0; ii < 4; ++ii) {
        int tl = ty + ii * 8;
        tile[tl][tx] = H[((size_t)(t0 + tl) * B_BATCH + b) * D_DIM + d0 + tx];
    }
    __syncthreads();
    #pragma unroll
    for (int ii = 0; ii < 4; ++ii) {
        int dl = ty + ii * 8;
        out[((size_t)b * D_DIM + d0 + dl) * T_SEQ + t0 + tx] = tile[tx][dl];
    }
}

// ---------------------------------------------------------------------------
// launch
// ---------------------------------------------------------------------------
extern "C" void kernel_launch(void* const* d_in, const int* in_sizes, int n_in,
                              void* d_out, int out_size, void* d_ws, size_t ws_size,
                              hipStream_t stream) {
    const float* x    = (const float*)d_in[0];   // [B, D, T]
    const float* W    = (const float*)d_in[1];   // [D, 3D]
    const float* U    = (const float*)d_in[2];   // [D, 3D]
    const float* bias = (const float*)d_in[3];   // [3D]
    float* out = (float*)d_out;                  // [B, D, T]

    const size_t FRAG_BYTES = (size_t)NTILES * KBLK * 32 * 16 * sizeof(unsigned short);
    const size_t GX_BYTES   = (size_t)T_SEQ * B_BATCH * N3 * sizeof(float);
    char* ws = (char*)d_ws;
    unsigned short* Wf = (unsigned short*)(ws);
    unsigned short* Uf = (unsigned short*)(ws + FRAG_BYTES);
    float*          GX = (float*)(ws + 2 * FRAG_BYTES);
    float*          H  = (float*)(ws + 2 * FRAG_BYTES + GX_BYTES);

    k0_swizzle_weights<<<dim3(192), 256, 0, stream>>>(W, U, Wf, Uf);
    k1_gx_gemm<<<dim3(T_SEQ / 16, B_BATCH), 128, 0, stream>>>(x, Wf, bias, GX);
    k2_recurrence<<<dim3(B_BATCH / 16), 256, 0, stream>>>(GX, Uf, H);
    k3_transpose<<<dim3(T_SEQ / 32, D_DIM / 32, B_BATCH), 256, 0, stream>>>(H, out);
}
